// AdditiveAttnActLayer_33792802685124
// MI455X (gfx1250) — compile-verified, matrix operand order checked by reading, not run on hardware
//
#include <hip/hip_runtime.h>
#include <hip/hip_bf16.h>

#define DIM   256
#define NHEAD 8
#define HDIM  32

typedef __attribute__((ext_vector_type(16))) __bf16 v16bf;
typedef __attribute__((ext_vector_type(8)))  __bf16 v8bf;
typedef __attribute__((ext_vector_type(8)))  float  v8f;
typedef __attribute__((__vector_size__(16))) int    v4i;

union FragBF { v16bf v; v8bf h[2]; };

// ---------------- elementwise fp32 -> bf16 ----------------
__global__ void f32_to_bf16_kernel(const float* __restrict__ in, __bf16* __restrict__ out, size_t n) {
  size_t i = (size_t)blockIdx.x * blockDim.x + threadIdx.x;
  size_t stride = (size_t)gridDim.x * blockDim.x;
  for (; i < n; i += stride) out[i] = (__bf16)in[i];
}

// bias for packed [Q;K;V] projection: bq for cols 0..255, zeros for K,V (no bias in ref)
__global__ void build_bias768_kernel(const float* __restrict__ bq, float* __restrict__ out) {
  int i = blockIdx.x * blockDim.x + threadIdx.x;
  if (i < 768) out[i] = (i < DIM) ? bq[i] : 0.0f;
}

// ---------------- bf16 WMMA GEMM:  C[M,Ncols] = A[M,K] @ W[Ncols,K]^T (+bias)(+res)(relu) ----
// Block (8 waves) owns one 64-wide N tile and a chunk of 32 M-tiles (4 per wave).
// Weight tile (64 x KDIM bf16) staged to LDS once per block via async-to-LDS (ASYNCcnt),
// padded stride KDIM+8 elements for bank-conflict-free ds_load_b128 fragment reads.
// Requires M % 16 == 0, Ncols % 64 == 0, KDIM % 32 == 0 (all true for this layer).
#define MT_CHUNK 32

template<int KDIM, bool RELU, bool OUT_BF16, bool HAS_RES>
__global__ __launch_bounds__(256)
void gemm_bf16_wmma(const __bf16* __restrict__ A, const __bf16* __restrict__ W,
                    const float* __restrict__ bias, const float* __restrict__ res,
                    float* __restrict__ outF, __bf16* __restrict__ outB,
                    int M, int Ncols)
{
  constexpr int TSTRIDE = KDIM + 8;   // +16B pad: lane stride 528B -> banks 4*l, conflict-free
  __shared__ __attribute__((aligned(16))) __bf16 tileW[64 * TSTRIDE];

  const int lane  = threadIdx.x & 31;
  const int wave  = threadIdx.x >> 5;
  const int ntN   = Ncols >> 6;
  const int mtiles = M >> 4;
  const int tn    = blockIdx.x % ntN;
  const int chunk = blockIdx.x / ntN;

  // ---- stage weight tile into LDS (async DMA path when available) ----
  {
    const __bf16* wsrc = W + (size_t)(tn * 64) * KDIM;
    constexpr int CHUNKS = 64 * (KDIM / 8);        // 16B chunks
    for (int c = threadIdx.x; c < CHUNKS; c += 256) {
      const int row  = c / (KDIM / 8);
      const int col8 = c % (KDIM / 8);
      const __bf16* g = wsrc  + (size_t)row * KDIM    + col8 * 8;
      __bf16*       l = tileW + (size_t)row * TSTRIDE + col8 * 8;
#if __has_builtin(__builtin_amdgcn_global_load_async_to_lds_b128)
      __builtin_amdgcn_global_load_async_to_lds_b128(
          (__attribute__((address_space(1))) v4i*)g,
          (__attribute__((address_space(3))) v4i*)l, 0, 0);
#else
      *(v8bf*)l = *(const v8bf*)g;
#endif
    }
#if __has_builtin(__builtin_amdgcn_global_load_async_to_lds_b128)
#if __has_builtin(__builtin_amdgcn_s_wait_asynccnt)
    __builtin_amdgcn_s_wait_asynccnt(0);
#else
    asm volatile("s_wait_asynccnt 0" ::: "memory");
#endif
#endif
    __syncthreads();
  }

  const int lr   = lane & 15;
  const int koff = (lane < 16) ? 0 : 8;        // ISA 16-bit A/B fragment K split per lane half
  const __bf16* bbase = tileW + (size_t)lr * TSTRIDE + koff;

  int mt_end = (chunk + 1) * MT_CHUNK;
  if (mt_end > mtiles) mt_end = mtiles;

  for (int mt = chunk * MT_CHUNK + wave; mt < mt_end; mt += 8) {
    const __bf16* aptr = A + (size_t)(mt * 16 + lr) * KDIM + koff;
    v8f acc[4] = {};
    #pragma unroll
    for (int k0 = 0; k0 < KDIM; k0 += 32) {
      FragBF a;
      a.h[0] = *(const v8bf*)(aptr + k0);
      a.h[1] = *(const v8bf*)(aptr + k0 + 16);
      __builtin_prefetch(aptr + k0 + 64, 0, 3);  // -> global_prefetch_b8 (A stream)
      #pragma unroll
      for (int j = 0; j < 4; ++j) {
        FragBF b;                                 // ds_load_b128 x2 from LDS tile
        b.h[0] = *(const v8bf*)(bbase + (size_t)j * 16 * TSTRIDE + k0);
        b.h[1] = *(const v8bf*)(bbase + (size_t)j * 16 * TSTRIDE + k0 + 16);
        acc[j] = __builtin_amdgcn_wmma_f32_16x16x32_bf16(
            false, a.v, false, b.v, (short)0, acc[j], false, false);
      }
    }
    // C layout: VGPR p -> (M = p | p+8 by lane half, N = lane&15)
    const int mbase = mt * 16 + ((lane < 16) ? 0 : 8);
    #pragma unroll
    for (int j = 0; j < 4; ++j) {
      const int n  = tn * 64 + j * 16 + lr;
      const float bv = bias ? bias[n] : 0.0f;
      #pragma unroll
      for (int p = 0; p < 8; ++p) {
        const size_t m = (size_t)(mbase + p);
        float vv = acc[j][p] + bv;
        if (HAS_RES) vv += res[m * (size_t)Ncols + n];
        if (RELU)    vv = fmaxf(vv, 0.0f);
        if (OUT_BF16) outB[m * (size_t)Ncols + n] = (__bf16)vv;
        else          outF[m * (size_t)Ncols + n] = vv;
      }
    }
  }
}

// ---------------- edge attention: conn + clipped score + exp + segment-sum ----------------
// One block per edge; wave w = head w (H==8 waves), lane = head-dim (HD==32 == wave32).
__global__ __launch_bounds__(256)
void edge_attn_kernel(const __bf16* __restrict__ nqkv, const __bf16* __restrict__ eq,
                      const int* __restrict__ src, const int* __restrict__ dst,
                      const float* __restrict__ Aw,
                      __bf16* __restrict__ conn, float* __restrict__ wexp,
                      float* __restrict__ ssum)
{
  const int e    = blockIdx.x;
  const int h    = threadIdx.x >> 5;
  const int lane = threadIdx.x & 31;
  const int d    = (h << 5) | lane;
  const int s = src[e], t = dst[e];
  float nk = (float)nqkv[(size_t)s * 768 + 256 + d];
  float nq = (float)nqkv[(size_t)t * 768 + d];
  float ev = (float)eq[(size_t)e * DIM + d];
  float c  = fmaxf(nk + nq + ev, 0.0f);            // relu
  conn[(size_t)e * DIM + d] = (__bf16)c;
  // score_h = sum_hd conn * Aw[hd,h,0]
  float p = c * Aw[lane * NHEAD + h];
  #pragma unroll
  for (int off = 16; off > 0; off >>= 1) p += __shfl_xor(p, off, 32);
  if (lane == 0) {
    p = fminf(fmaxf(p, -5.0f), 5.0f);              // clip: |s|<=5 => exp safe w/o max-sub
    float w = __expf(p);
    wexp[(size_t)e * NHEAD + h] = w;
    atomicAdd(&ssum[(size_t)t * NHEAD + h], w);
  }
}

// ---------------- edge aggregation: scatter agg1 += Nv[src]*wn, agg2 += conn*wn ----------------
__global__ __launch_bounds__(256)
void edge_agg_kernel(const __bf16* __restrict__ nqkv, const __bf16* __restrict__ conn,
                     const int* __restrict__ src, const int* __restrict__ dst,
                     const float* __restrict__ wexp, const float* __restrict__ ssum,
                     float* __restrict__ agg1, float* __restrict__ agg2)
{
  const int e    = blockIdx.x;
  const int h    = threadIdx.x >> 5;
  const int lane = threadIdx.x & 31;
  const int d    = (h << 5) | lane;
  const int s = src[e], t = dst[e];
  float wn = wexp[(size_t)e * NHEAD + h] / (ssum[(size_t)t * NHEAD + h] + 1e-16f);
  float nv = (float)nqkv[(size_t)s * 768 + 512 + d];
  float cv = (float)conn[(size_t)e * DIM + d];
  atomicAdd(&agg1[(size_t)t * DIM + d], nv * wn);
  atomicAdd(&agg2[(size_t)t * DIM + d], cv * wn);
}

// ---------------- node post: agg = agg1 + agg2@Ew (per-head 32x32), degree scaling -> bf16 ----
__global__ __launch_bounds__(256)
void node_post_kernel(const float* __restrict__ agg1, const float* __restrict__ agg2,
                      const float* __restrict__ Ew, const float* __restrict__ deg_coef,
                      const float* __restrict__ log_deg, __bf16* __restrict__ nhs)
{
  const int n    = blockIdx.x;
  const int h    = threadIdx.x >> 5;
  const int lane = threadIdx.x & 31;
  const int d    = (h << 5) | lane;
  float a2 = agg2[(size_t)n * DIM + d];
  float acc = 0.0f;
  #pragma unroll
  for (int dp = 0; dp < HDIM; ++dp) {
    float v = __shfl(a2, dp, 32);                       // broadcast agg2[n,h,dp]
    acc += v * Ew[(dp * NHEAD + h) * HDIM + lane];      // Ew[dp,h,lane]
  }
  float nh = agg1[(size_t)n * DIM + d] + acc;
  float ld = log_deg[n];
  nh = nh * deg_coef[2 * d] + nh * ld * deg_coef[2 * d + 1];
  nhs[(size_t)n * DIM + d] = (__bf16)nh;
}

// ---------------- batchnorm: column stats (sum, sumsq) ----------------
__global__ __launch_bounds__(256)
void bn_stats_kernel(const float* __restrict__ in, float* __restrict__ stats,
                     int M, int rowsPerBlock)
{
  const int c = threadIdx.x;                 // column 0..255
  size_t r0 = (size_t)blockIdx.x * rowsPerBlock;
  size_t r1 = r0 + rowsPerBlock; if (r1 > (size_t)M) r1 = (size_t)M;
  float s = 0.0f, s2 = 0.0f;
  for (size_t r = r0; r < r1; ++r) {
    float v = in[r * DIM + c];
    s += v; s2 += v * v;
  }
  atomicAdd(&stats[c], s);
  atomicAdd(&stats[DIM + c], s2);
}

// ---------------- batchnorm apply (biased var, eps=1e-5), optional relu / bf16 copy ----------
__global__ void bn_apply_kernel(const float* __restrict__ in, const float* __restrict__ stats,
                                const float* __restrict__ g, const float* __restrict__ b,
                                float* __restrict__ outF, __bf16* __restrict__ outB,
                                size_t total, float invM, int relu)
{
  size_t i = (size_t)blockIdx.x * blockDim.x + threadIdx.x;
  size_t stride = (size_t)gridDim.x * blockDim.x;
  for (; i < total; i += stride) {
    int c = (int)(i & (DIM - 1));
    float mu  = stats[c] * invM;
    float var = stats[DIM + c] * invM - mu * mu;
    float sc  = rsqrtf(var + 1e-5f);
    float v = (in[i] - mu) * sc * g[c] + b[c];
    if (relu) v = fmaxf(v, 0.0f);
    if (outF) outF[i] = v;
    if (outB) outB[i] = (__bf16)v;
  }
}

static inline size_t smin_sz(size_t a, size_t b) { return a < b ? a : b; }

extern "C" void kernel_launch(void* const* d_in, const int* in_sizes, int n_in,
                              void* d_out, int out_size, void* d_ws, size_t ws_size,
                              hipStream_t stream)
{
  (void)n_in; (void)out_size; (void)ws_size;
  const float* x         = (const float*)d_in[0];
  const float* edge_attr = (const float*)d_in[1];
  const float* log_deg   = (const float*)d_in[2];
  const int*   eidx      = (const int*)  d_in[3];
  const float* Wq  = (const float*)d_in[4];
  const float* bq  = (const float*)d_in[5];
  const float* Wk  = (const float*)d_in[6];
  const float* Wv  = (const float*)d_in[7];
  const float* WEq = (const float*)d_in[8];
  const float* bEq = (const float*)d_in[9];
  const float* Aw  = (const float*)d_in[10];
  const float* Ew  = (const float*)d_in[11];
  const float* deg_coef = (const float*)d_in[12];
  const float* WNo = (const float*)d_in[13];
  const float* bNo = (const float*)d_in[14];
  const float* WEo = (const float*)d_in[15];
  const float* bEo = (const float*)d_in[16];
  const float* g1n = (const float*)d_in[17];
  const float* b1n = (const float*)d_in[18];
  const float* g1e = (const float*)d_in[19];
  const float* b1e = (const float*)d_in[20];
  const float* W1  = (const float*)d_in[21];
  const float* bb1 = (const float*)d_in[22];
  const float* W2  = (const float*)d_in[23];
  const float* bb2 = (const float*)d_in[24];
  const float* g2  = (const float*)d_in[25];
  const float* b2  = (const float*)d_in[26];

  const int N = in_sizes[0] / DIM;   // 50000 (multiple of 16)
  const int E = in_sizes[1] / DIM;   // 400000 (multiple of 16)
  const int* srcI = eidx;
  const int* dstI = eidx + E;

  float* out_nh = (float*)d_out;                    // [N,256]
  float* out_eh = out_nh + (size_t)N * DIM;         // [E,256]

  // ---- carve workspace ----
  char* ws = (char*)d_ws;
  size_t off = 0;
  auto alloc = [&](size_t bytes) -> void* {
    off = (off + 255) & ~(size_t)255;
    void* p = ws + off;
    off += bytes;
    return p;
  };
  __bf16* x_bf    = (__bf16*)alloc((size_t)N * DIM * 2);
  __bf16* ea_bf   = (__bf16*)alloc((size_t)E * DIM * 2);
  __bf16* wqkv_bf = (__bf16*)alloc((size_t)768 * DIM * 2);
  __bf16* weq_bf  = (__bf16*)alloc((size_t)DIM * DIM * 2);
  __bf16* wno_bf  = (__bf16*)alloc((size_t)DIM * DIM * 2);
  __bf16* weo_bf  = (__bf16*)alloc((size_t)DIM * DIM * 2);
  __bf16* w1_bf   = (__bf16*)alloc((size_t)512 * DIM * 2);
  __bf16* w2_bf   = (__bf16*)alloc((size_t)DIM * 512 * 2);
  float*  bias768 = (float*) alloc(768 * 4);
  __bf16* nqkv_bf = (__bf16*)alloc((size_t)N * 768 * 2);
  __bf16* eq_bf   = (__bf16*)alloc((size_t)E * DIM * 2);
  __bf16* conn_bf = (__bf16*)alloc((size_t)E * DIM * 2);
  float*  wexp    = (float*) alloc((size_t)E * NHEAD * 4);
  float*  ssum    = (float*) alloc((size_t)N * NHEAD * 4);
  float*  agg1    = (float*) alloc((size_t)N * DIM * 4);
  float*  agg2    = (float*) alloc((size_t)N * DIM * 4);
  __bf16* nhs_bf  = (__bf16*)alloc((size_t)N * DIM * 2);
  float*  nh_pre  = (float*) alloc((size_t)N * DIM * 4);
  float*  nhbn_f  = (float*) alloc((size_t)N * DIM * 4);
  __bf16* nhbn_bf = (__bf16*)alloc((size_t)N * DIM * 2);
  __bf16* ffn_bf  = (__bf16*)alloc((size_t)N * 512 * 2);
  float*  nh2_pre = (float*) alloc((size_t)N * DIM * 4);
  float*  statsN1 = (float*) alloc(2 * DIM * 4);
  float*  statsE  = (float*) alloc(2 * DIM * 4);
  float*  statsN2 = (float*) alloc(2 * DIM * 4);

  auto conv = [&](const float* s, __bf16* d, size_t n) {
    int blocks = (int)smin_sz((n + 255) / 256, (size_t)131072);
    f32_to_bf16_kernel<<<blocks, 256, 0, stream>>>(s, d, n);
  };

  // ---- stage 0: precision conversion + init ----
  conv(x,         x_bf,  (size_t)N * DIM);
  conv(edge_attr, ea_bf, (size_t)E * DIM);
  conv(Wq,  wqkv_bf,               (size_t)DIM * DIM);
  conv(Wk,  wqkv_bf + 256 * DIM,   (size_t)DIM * DIM);
  conv(Wv,  wqkv_bf + 512 * DIM,   (size_t)DIM * DIM);
  conv(WEq, weq_bf, (size_t)DIM * DIM);
  conv(WNo, wno_bf, (size_t)DIM * DIM);
  conv(WEo, weo_bf, (size_t)DIM * DIM);
  conv(W1,  w1_bf,  (size_t)512 * DIM);
  conv(W2,  w2_bf,  (size_t)DIM * 512);
  build_bias768_kernel<<<3, 256, 0, stream>>>(bq, bias768);
  (void)hipMemsetAsync(ssum,    0, (size_t)N * NHEAD * 4, stream);
  (void)hipMemsetAsync(agg1,    0, (size_t)N * DIM * 4,   stream);
  (void)hipMemsetAsync(agg2,    0, (size_t)N * DIM * 4,   stream);
  (void)hipMemsetAsync(statsN1, 0, 2 * DIM * 4, stream);
  (void)hipMemsetAsync(statsE,  0, 2 * DIM * 4, stream);
  (void)hipMemsetAsync(statsN2, 0, 2 * DIM * 4, stream);

  auto gemm_blocks = [](int M, int Ncols) {
    int mt = M / 16;
    return (Ncols / 64) * ((mt + MT_CHUNK - 1) / MT_CHUNK);
  };

  // ---- stage 1: projections (WMMA bf16, LDS-staged weights) ----
  gemm_bf16_wmma<256, false, true, false><<<gemm_blocks(N, 768), 256, 0, stream>>>(
      x_bf, wqkv_bf, bias768, nullptr, nullptr, nqkv_bf, N, 768);
  gemm_bf16_wmma<256, false, true, false><<<gemm_blocks(E, 256), 256, 0, stream>>>(
      ea_bf, weq_bf, bEq, nullptr, nullptr, eq_bf, E, 256);

  // ---- stage 2: edge attention + softmax denominators ----
  edge_attn_kernel<<<E, 256, 0, stream>>>(nqkv_bf, eq_bf, srcI, dstI, Aw, conn_bf, wexp, ssum);

  // ---- stage 3: normalized aggregation (L2-resident scatter atomics) ----
  edge_agg_kernel<<<E, 256, 0, stream>>>(nqkv_bf, conn_bf, srcI, dstI, wexp, ssum, agg1, agg2);

  // ---- stage 4: per-head Ew mix + degree scaling ----
  node_post_kernel<<<N, 256, 0, stream>>>(agg1, agg2, Ew, deg_coef, log_deg, nhs_bf);

  // ---- stage 5: output projections with residuals (WMMA) ----
  gemm_bf16_wmma<256, false, false, true><<<gemm_blocks(N, 256), 256, 0, stream>>>(
      nhs_bf, wno_bf, bNo, x, nh_pre, nullptr, N, 256);
  gemm_bf16_wmma<256, false, false, true><<<gemm_blocks(E, 256), 256, 0, stream>>>(
      conn_bf, weo_bf, bEo, edge_attr, out_eh, nullptr, E, 256);

  // ---- stage 6: batchnorm 1 (nodes: f32 + bf16 copies; edges: in-place + relu -> final eh) ----
  bn_stats_kernel<<<(N + 511) / 512, 256, 0, stream>>>(nh_pre, statsN1, N, 512);
  {
    size_t tot = (size_t)N * DIM;
    int blocks = (int)smin_sz((tot + 255) / 256, (size_t)131072);
    bn_apply_kernel<<<blocks, 256, 0, stream>>>(nh_pre, statsN1, g1n, b1n,
                                                nhbn_f, nhbn_bf, tot, 1.0f / (float)N, 0);
  }
  bn_stats_kernel<<<(E + 511) / 512, 256, 0, stream>>>(out_eh, statsE, E, 512);
  {
    size_t tot = (size_t)E * DIM;
    int blocks = (int)smin_sz((tot + 255) / 256, (size_t)131072);
    bn_apply_kernel<<<blocks, 256, 0, stream>>>(out_eh, statsE, g1e, b1e,
                                                out_eh, nullptr, tot, 1.0f / (float)E, 1);
  }

  // ---- stage 7: node FFN (WMMA) ----
  gemm_bf16_wmma<256, true, true, false><<<gemm_blocks(N, 512), 256, 0, stream>>>(
      nhbn_bf, w1_bf, bb1, nullptr, nullptr, ffn_bf, N, 512);
  gemm_bf16_wmma<512, false, false, true><<<gemm_blocks(N, 256), 256, 0, stream>>>(
      ffn_bf, w2_bf, bb2, nhbn_f, nh2_pre, nullptr, N, 256);

  // ---- stage 8: batchnorm 2 + relu -> final nh ----
  bn_stats_kernel<<<(N + 511) / 512, 256, 0, stream>>>(nh2_pre, statsN2, N, 512);
  {
    size_t tot = (size_t)N * DIM;
    int blocks = (int)smin_sz((tot + 255) / 256, (size_t)131072);
    bn_apply_kernel<<<blocks, 256, 0, stream>>>(nh2_pre, statsN2, g2, b2,
                                                out_nh, nullptr, tot, 1.0f / (float)N, 1);
  }
}
